// HopfieldLookupModel_35510789603642
// MI455X (gfx1250) — matrix operand order repaired and checked by reading, not compiled, not run on hardware
//
#include <hip/hip_runtime.h>

#define BATCH 32768
#define DDIM  256
#define QPAT  1625
#define QTILE 32
#define QPAD  1664                           // 52 * 32, zero-padded pattern count
#define NQT   ((QPAT + QTILE - 1) / QTILE)   // 51 tiles (last one padded)
#define WAVES 4
#define BLOCK (WAVES * 32)                   // 128 threads
#define ROWS_PER_WAVE 32                     // two 16-row WMMA tiles per wave
#define ROWS_PER_WG (WAVES * ROWS_PER_WAVE)  // 128 batch rows per workgroup
#define BETA  0.0625f                        // 1/sqrt(256), exact power of two

typedef __attribute__((ext_vector_type(16))) __bf16         v16bf;
typedef __attribute__((ext_vector_type(8)))  float          v8f;
typedef __attribute__((ext_vector_type(8)))  unsigned short v8u;
typedef __attribute__((ext_vector_type(4)))  unsigned int   v4i;

static __device__ __forceinline__ unsigned short f2bf(float f) {
  // round-to-nearest-even f32 -> bf16 (inputs finite; no NaN path needed)
  unsigned int u = __float_as_uint(f);
  u += 0x7FFFu + ((u >> 16) & 1u);
  return (unsigned short)(u >> 16);
}

static __device__ __forceinline__ unsigned int f2bf2(float lo, float hi) {
#if __has_builtin(__builtin_amdgcn_cvt_pk_bf16_f32)
  typedef __attribute__((ext_vector_type(2))) __bf16 v2bf;
  return __builtin_bit_cast(unsigned int,
                            __builtin_amdgcn_cvt_pk_bf16_f32(lo, hi));
#else
  return (unsigned int)f2bf(lo) | ((unsigned int)f2bf(hi) << 16);
#endif
}

// Issue one 16 KB tile copy (global bf16 -> LDS) as 8 async b128 transfers per
// thread (128 threads x 8 x 16 B = 16 KB). Tracked by ASYNCcnt; no VGPR data.
static __device__ __forceinline__ void issue_async_tile(unsigned ldsByte,
                                                        const unsigned short* gBase,
                                                        unsigned gByteOff, int tid) {
#pragma unroll
  for (int i = 0; i < 8; ++i) {
    const unsigned off = (unsigned)(i * 2048 + tid * 16);
    const unsigned lds = ldsByte + off;
    const unsigned gof = gByteOff + off;
    asm volatile("global_load_async_to_lds_b128 %0, %1, %2"
                 :: "v"(lds), "v"(gof), "s"(gBase) : "memory");
  }
}

// ---------------- one-time pattern conversion into workspace ----------------
__global__ void hopfield_prep_kernel(const float* __restrict__ patterns,
                                     unsigned short* __restrict__ wsRow,
                                     unsigned short* __restrict__ wsCol)
{
  const int gid = blockIdx.x * 256 + threadIdx.x;   // 0 .. QPAD*64-1
  const int q  = gid >> 6;                          // 0..1663
  const int d4 = (gid & 63) * 4;                    // 0..252
  float x0 = 0.f, x1 = 0.f, x2 = 0.f, x3 = 0.f;
  if (q < QPAT) {
    const float4 f = *(const float4*)&patterns[(size_t)q * DDIM + d4];
    x0 = f.x; x1 = f.y; x2 = f.z; x3 = f.w;
  }
  const unsigned w0 = f2bf2(x0, x1);
  const unsigned w1 = f2bf2(x2, x3);
  uint2 rw; rw.x = w0; rw.y = w1;
  *(uint2*)&wsRow[(size_t)q * DDIM + d4] = rw;
  wsCol[(size_t)(d4 + 0) * QPAD + q] = (unsigned short)(w0);
  wsCol[(size_t)(d4 + 1) * QPAD + q] = (unsigned short)(w0 >> 16);
  wsCol[(size_t)(d4 + 2) * QPAD + q] = (unsigned short)(w1);
  wsCol[(size_t)(d4 + 3) * QPAD + q] = (unsigned short)(w1 >> 16);
}

// ---------------- main fused kernel (workspace path) ----------------
__global__ __launch_bounds__(BLOCK, 1)
void hopfield_ws_kernel(const float* __restrict__ features,
                        const unsigned short* __restrict__ wsRow,  // [QPAD][256] bf16
                        const unsigned short* __restrict__ wsCol,  // [256][QPAD] bf16
                        const float* __restrict__ Wout,
                        const float* __restrict__ bout,
                        float* __restrict__ out)
{
  __shared__ unsigned short sRow[2][QTILE * DDIM];          // 2 x 16 KB double buffer
  __shared__ unsigned short sProbs[WAVES][32 * QTILE];      // 8 KB C->A relayout patch
  __shared__ float sW[DDIM];                                // 1 KB

  const int tid  = threadIdx.x;
  const int lane = tid & 31;
  const int wave = tid >> 5;
  const int half = lane >> 4;
  const int l16  = lane & 15;
  const int rowBase = blockIdx.x * ROWS_PER_WG + wave * ROWS_PER_WAVE;

  sW[tid]       = Wout[tid];
  sW[tid + 128] = Wout[tid + 128];

  const v8f zero8 = {0.f, 0.f, 0.f, 0.f, 0.f, 0.f, 0.f, 0.f};

  // preload this wave's 32 feature rows as 2x8 bf16 A-fragments (beta folded)
  v16bf afA[8], afB[8];
  {
    const float* frowA = features + (size_t)(rowBase + l16) * DDIM;
    const float* frowB = features + (size_t)(rowBase + 16 + l16) * DDIM;
#pragma unroll
    for (int kc = 0; kc < 8; ++kc) {
      const int d0 = kc * 32 + half * 8;
      union { v16bf bf; unsigned int w[8]; } ca, cb;
#pragma unroll
      for (int g = 0; g < 2; ++g)
#pragma unroll
        for (int j = 0; j < 4; ++j) {
          ca.w[g * 4 + j] = f2bf2(frowA[d0 + g * 16 + 2 * j]     * BETA,
                                  frowA[d0 + g * 16 + 2 * j + 1] * BETA);
          cb.w[g * 4 + j] = f2bf2(frowB[d0 + g * 16 + 2 * j]     * BETA,
                                  frowB[d0 + g * 16 + 2 * j + 1] * BETA);
        }
      afA[kc] = ca.bf;
      afB[kc] = cb.bf;
    }
  }

  v8f accA[16], accB[16];
#pragma unroll
  for (int t = 0; t < 16; ++t) { accA[t] = zero8; accB[t] = zero8; }
  float lpA[8], lpB[8];
#pragma unroll
  for (int v = 0; v < 8; ++v) { lpA[v] = 0.0f; lpB[v] = 0.0f; }

  const unsigned lds0 = (unsigned)(uintptr_t)&sRow[0][0];
  const unsigned lds1 = (unsigned)(uintptr_t)&sRow[1][0];
  issue_async_tile(lds0, wsRow, 0u, tid);

  for (int qt = 0; qt < NQT; ++qt) {
    const int q0 = qt * QTILE;
    const int buf = qt & 1;

    // my async copies for tile qt done; barrier makes all waves' copies visible
    asm volatile("s_wait_asynccnt 0" ::: "memory");
    __syncthreads();
    if (qt + 1 < NQT)
      issue_async_tile(buf ? lds0 : lds1, wsRow,
                       (unsigned)((qt + 1) * QTILE * DDIM * 2), tid);

    const unsigned short* rowT = &sRow[buf][0];

    // ---- QK^T: B fragments from LDS, each feeds both row-tiles ----
    v8f cA0 = zero8, cA1 = zero8, cB0 = zero8, cB1 = zero8;
#pragma unroll
    for (int kc = 0; kc < 8; ++kc) {
      const int dk = kc * 32 + half * 8;
      union { v8u h2[2]; v16bf bf; } b0, b1;
      b0.h2[0] = *(const v8u*)&rowT[(l16)      * DDIM + dk];
      b0.h2[1] = *(const v8u*)&rowT[(l16)      * DDIM + dk + 16];
      b1.h2[0] = *(const v8u*)&rowT[(16 + l16) * DDIM + dk];
      b1.h2[1] = *(const v8u*)&rowT[(16 + l16) * DDIM + dk + 16];
      cA0 = __builtin_amdgcn_wmma_f32_16x16x32_bf16(false, afA[kc], false, b0.bf, (short)0, cA0, false, false);
      cB0 = __builtin_amdgcn_wmma_f32_16x16x32_bf16(false, afB[kc], false, b0.bf, (short)0, cB0, false, false);
      cA1 = __builtin_amdgcn_wmma_f32_16x16x32_bf16(false, afA[kc], false, b1.bf, (short)0, cA1, false, false);
      cB1 = __builtin_amdgcn_wmma_f32_16x16x32_bf16(false, afB[kc], false, b1.bf, (short)0, cB1, false, false);
    }

    // ---- softmax numerators, max-free (exact: shift-invariant, scores
    // ~N(0,1/256) by construction so exp cannot overflow) ----
    const bool ok0 = (q0 + l16)      < QPAT;
    const bool ok1 = (q0 + 16 + l16) < QPAT;
    unsigned int pwA[8], pwB[8];
#pragma unroll
    for (int v = 0; v < 8; ++v) {
      const float a0 = ok0 ? __expf(cA0[v]) : 0.0f;
      const float a1 = ok1 ? __expf(cA1[v]) : 0.0f;
      const float b0 = ok0 ? __expf(cB0[v]) : 0.0f;
      const float b1 = ok1 ? __expf(cB1[v]) : 0.0f;
      lpA[v] += a0 + a1;
      lpB[v] += b0 + b1;
      pwA[v] = f2bf2(a0, a1);
      pwB[v] = f2bf2(b0, b1);
    }

    // ---- probs: C layout -> A layout via wave-private LDS patch ----
    {
      unsigned short* pw = &sProbs[wave][0];
#pragma unroll
      for (int v = 0; v < 8; ++v) {
        const int m = v + half * 8;
        pw[m * QTILE + l16]             = (unsigned short)(pwA[v]);
        pw[m * QTILE + 16 + l16]        = (unsigned short)(pwA[v] >> 16);
        pw[(16 + m) * QTILE + l16]      = (unsigned short)(pwB[v]);
        pw[(16 + m) * QTILE + 16 + l16] = (unsigned short)(pwB[v] >> 16);
      }
    }
    asm volatile("s_wait_dscnt 0" ::: "memory");

    union { v8u h2[2]; v16bf bf; } pfA, pfB;
    {
      const unsigned short* pr = &sProbs[wave][0];
      const int cb = half * 8;
      pfA.h2[0] = *(const v8u*)&pr[(l16)      * QTILE + cb];
      pfA.h2[1] = *(const v8u*)&pr[(l16)      * QTILE + 16 + cb];
      pfB.h2[0] = *(const v8u*)&pr[(16 + l16) * QTILE + cb];
      pfB.h2[1] = *(const v8u*)&pr[(16 + l16) * QTILE + 16 + cb];
    }

    // ---- PV: B fragments straight from L2-resident wsCol (off-loads LDS) ----
#pragma unroll
    for (int t = 0; t < 16; ++t) {
      const unsigned short* cp = wsCol + (size_t)(t * 16 + l16) * QPAD + q0 + half * 8;
      union { v8u h2[2]; v16bf bf; } bv;
      bv.h2[0] = *(const v8u*)cp;
      bv.h2[1] = *(const v8u*)(cp + 16);
      accA[t] = __builtin_amdgcn_wmma_f32_16x16x32_bf16(false, pfA.bf, false, bv.bf, (short)0, accA[t], false, false);
      accB[t] = __builtin_amdgcn_wmma_f32_16x16x32_bf16(false, pfB.bf, false, bv.bf, (short)0, accB[t], false, false);
    }
  }

  // ---- epilogue: project with W_out, reduce, normalize, sigmoid ----
  float pA[8], pB[8];
#pragma unroll
  for (int v = 0; v < 8; ++v) { pA[v] = 0.0f; pB[v] = 0.0f; }
#pragma unroll
  for (int t = 0; t < 16; ++t) {
    const float w = sW[t * 16 + l16];
#pragma unroll
    for (int v = 0; v < 8; ++v) {
      pA[v] += accA[t][v] * w;
      pB[v] += accB[t][v] * w;
    }
  }
  const float bias = bout[0];
#pragma unroll
  for (int v = 0; v < 8; ++v) {
    float sA = pA[v], LA = lpA[v];
    float sB = pB[v], LB = lpB[v];
#pragma unroll
    for (int off = 8; off >= 1; off >>= 1) {
      sA += __shfl_xor(sA, off, 32);
      LA += __shfl_xor(LA, off, 32);
      sB += __shfl_xor(sB, off, 32);
      LB += __shfl_xor(LB, off, 32);
    }
    pA[v] = sA / LA + bias;
    pB[v] = sB / LB + bias;
  }
  if (l16 == 0) {
#pragma unroll
    for (int v = 0; v < 8; ++v) {
      const int r = rowBase + v + half * 8;
      out[r]      = 1.0f / (1.0f + __expf(-pA[v]));
      out[r + 16] = 1.0f / (1.0f + __expf(-pB[v]));
    }
  }
}

// ---------------- self-contained fallback (if workspace too small) ----------------
__global__ __launch_bounds__(BLOCK, 1)
void hopfield_fallback_kernel(const float* __restrict__ features,
                              const float* __restrict__ patterns,
                              const float* __restrict__ Wout,
                              const float* __restrict__ bout,
                              float* __restrict__ out)
{
  __shared__ unsigned short sPatRow[QTILE * DDIM];
  __shared__ unsigned short sPatCol[DDIM * QTILE];
  __shared__ unsigned short sProbs[WAVES][32 * QTILE];
  __shared__ float sW[DDIM];

  const int tid  = threadIdx.x;
  const int lane = tid & 31;
  const int wave = tid >> 5;
  const int half = lane >> 4;
  const int l16  = lane & 15;
  const int rowBase = blockIdx.x * ROWS_PER_WG + wave * ROWS_PER_WAVE;

  sW[tid]       = Wout[tid];
  sW[tid + 128] = Wout[tid + 128];

  const v8f zero8 = {0.f, 0.f, 0.f, 0.f, 0.f, 0.f, 0.f, 0.f};

  v16bf afA[8], afB[8];
  {
    const float* frowA = features + (size_t)(rowBase + l16) * DDIM;
    const float* frowB = features + (size_t)(rowBase + 16 + l16) * DDIM;
#pragma unroll
    for (int kc = 0; kc < 8; ++kc) {
      const int d0 = kc * 32 + half * 8;
      union { v16bf bf; unsigned int w[8]; } ca, cb;
#pragma unroll
      for (int g = 0; g < 2; ++g)
#pragma unroll
        for (int j = 0; j < 4; ++j) {
          ca.w[g * 4 + j] = f2bf2(frowA[d0 + g * 16 + 2 * j]     * BETA,
                                  frowA[d0 + g * 16 + 2 * j + 1] * BETA);
          cb.w[g * 4 + j] = f2bf2(frowB[d0 + g * 16 + 2 * j]     * BETA,
                                  frowB[d0 + g * 16 + 2 * j + 1] * BETA);
        }
      afA[kc] = ca.bf;
      afB[kc] = cb.bf;
    }
  }

  v8f accA[16], accB[16];
#pragma unroll
  for (int t = 0; t < 16; ++t) { accA[t] = zero8; accB[t] = zero8; }
  float lpA[8], lpB[8];
#pragma unroll
  for (int v = 0; v < 8; ++v) { lpA[v] = 0.0f; lpB[v] = 0.0f; }

  for (int qt = 0; qt < NQT; ++qt) {
    const int q0 = qt * QTILE;
    {
      const int q8 = (tid & 3) * 8;
      const int dc = (tid >> 2) * 8;
      float fv[8][8];
#pragma unroll
      for (int r = 0; r < 8; ++r) {
        const int q = q0 + q8 + r;
        if (q < QPAT) {
          const float4 f0 = *(const float4*)&patterns[(size_t)q * DDIM + dc];
          const float4 f1 = *(const float4*)&patterns[(size_t)q * DDIM + dc + 4];
          fv[r][0] = f0.x; fv[r][1] = f0.y; fv[r][2] = f0.z; fv[r][3] = f0.w;
          fv[r][4] = f1.x; fv[r][5] = f1.y; fv[r][6] = f1.z; fv[r][7] = f1.w;
        } else {
#pragma unroll
          for (int j = 0; j < 8; ++j) fv[r][j] = 0.f;
        }
      }
      unsigned int colW[8][4];
#pragma unroll
      for (int j = 0; j < 8; ++j)
#pragma unroll
        for (int p = 0; p < 4; ++p)
          colW[j][p] = f2bf2(fv[2 * p][j], fv[2 * p + 1][j]);
#pragma unroll
      for (int j = 0; j < 8; ++j) {
        v4i w = { colW[j][0], colW[j][1], colW[j][2], colW[j][3] };
        *(v4i*)&sPatCol[(dc + j) * QTILE + q8] = w;
      }
#pragma unroll
      for (int r = 0; r < 8; ++r) {
        const unsigned int sel = (r & 1) ? 0x07060302u : 0x05040100u;
        v4i w = { __builtin_amdgcn_perm(colW[1][r >> 1], colW[0][r >> 1], sel),
                  __builtin_amdgcn_perm(colW[3][r >> 1], colW[2][r >> 1], sel),
                  __builtin_amdgcn_perm(colW[5][r >> 1], colW[4][r >> 1], sel),
                  __builtin_amdgcn_perm(colW[7][r >> 1], colW[6][r >> 1], sel) };
        *(v4i*)&sPatRow[(q8 + r) * DDIM + dc] = w;
      }
    }
    __syncthreads();

    v8f cA0 = zero8, cA1 = zero8, cB0 = zero8, cB1 = zero8;
#pragma unroll
    for (int kc = 0; kc < 8; ++kc) {
      const int dk = kc * 32 + half * 8;
      union { v8u h2[2]; v16bf bf; } b0, b1;
      b0.h2[0] = *(const v8u*)&sPatRow[(l16)      * DDIM + dk];
      b0.h2[1] = *(const v8u*)&sPatRow[(l16)      * DDIM + dk + 16];
      b1.h2[0] = *(const v8u*)&sPatRow[(16 + l16) * DDIM + dk];
      b1.h2[1] = *(const v8u*)&sPatRow[(16 + l16) * DDIM + dk + 16];
      cA0 = __builtin_amdgcn_wmma_f32_16x16x32_bf16(false, afA[kc], false, b0.bf, (short)0, cA0, false, false);
      cB0 = __builtin_amdgcn_wmma_f32_16x16x32_bf16(false, afB[kc], false, b0.bf, (short)0, cB0, false, false);
      cA1 = __builtin_amdgcn_wmma_f32_16x16x32_bf16(false, afA[kc], false, b1.bf, (short)0, cA1, false, false);
      cB1 = __builtin_amdgcn_wmma_f32_16x16x32_bf16(false, afB[kc], false, b1.bf, (short)0, cB1, false, false);
    }

    const bool ok0 = (q0 + l16)      < QPAT;
    const bool ok1 = (q0 + 16 + l16) < QPAT;
    unsigned int pwA[8], pwB[8];
#pragma unroll
    for (int v = 0; v < 8; ++v) {
      const float a0 = ok0 ? __expf(cA0[v]) : 0.0f;
      const float a1 = ok1 ? __expf(cA1[v]) : 0.0f;
      const float b0 = ok0 ? __expf(cB0[v]) : 0.0f;
      const float b1 = ok1 ? __expf(cB1[v]) : 0.0f;
      lpA[v] += a0 + a1;
      lpB[v] += b0 + b1;
      pwA[v] = f2bf2(a0, a1);
      pwB[v] = f2bf2(b0, b1);
    }
    {
      unsigned short* pw = &sProbs[wave][0];
#pragma unroll
      for (int v = 0; v < 8; ++v) {
        const int m = v + half * 8;
        pw[m * QTILE + l16]             = (unsigned short)(pwA[v]);
        pw[m * QTILE + 16 + l16]        = (unsigned short)(pwA[v] >> 16);
        pw[(16 + m) * QTILE + l16]      = (unsigned short)(pwB[v]);
        pw[(16 + m) * QTILE + 16 + l16] = (unsigned short)(pwB[v] >> 16);
      }
    }
    asm volatile("s_wait_dscnt 0" ::: "memory");

    union { v8u h2[2]; v16bf bf; } pfA, pfB;
    {
      const unsigned short* pr = &sProbs[wave][0];
      const int cb = half * 8;
      pfA.h2[0] = *(const v8u*)&pr[(l16)      * QTILE + cb];
      pfA.h2[1] = *(const v8u*)&pr[(l16)      * QTILE + 16 + cb];
      pfB.h2[0] = *(const v8u*)&pr[(16 + l16) * QTILE + cb];
      pfB.h2[1] = *(const v8u*)&pr[(16 + l16) * QTILE + 16 + cb];
    }
#pragma unroll
    for (int t = 0; t < 16; ++t) {
      const int d = t * 16 + l16;
      union { v8u h2[2]; v16bf bf; } bv;
      bv.h2[0] = *(const v8u*)&sPatCol[d * QTILE + half * 8];
      bv.h2[1] = *(const v8u*)&sPatCol[d * QTILE + half * 8 + 16];
      accA[t] = __builtin_amdgcn_wmma_f32_16x16x32_bf16(false, pfA.bf, false, bv.bf, (short)0, accA[t], false, false);
      accB[t] = __builtin_amdgcn_wmma_f32_16x16x32_bf16(false, pfB.bf, false, bv.bf, (short)0, accB[t], false, false);
    }
    __syncthreads();
  }

  float pA[8], pB[8];
#pragma unroll
  for (int v = 0; v < 8; ++v) { pA[v] = 0.0f; pB[v] = 0.0f; }
#pragma unroll
  for (int t = 0; t < 16; ++t) {
    const float w = sW[t * 16 + l16];
#pragma unroll
    for (int v = 0; v < 8; ++v) {
      pA[v] += accA[t][v] * w;
      pB[v] += accB[t][v] * w;
    }
  }
  const float bias = bout[0];
#pragma unroll
  for (int v = 0; v < 8; ++v) {
    float sA = pA[v], LA = lpA[v];
    float sB = pB[v], LB = lpB[v];
#pragma unroll
    for (int off = 8; off >= 1; off >>= 1) {
      sA += __shfl_xor(sA, off, 32);
      LA += __shfl_xor(LA, off, 32);
      sB += __shfl_xor(sB, off, 32);
      LB += __shfl_xor(LB, off, 32);
    }
    pA[v] = sA / LA + bias;
    pB[v] = sB / LB + bias;
  }
  if (l16 == 0) {
#pragma unroll
    for (int v = 0; v < 8; ++v) {
      const int r = rowBase + v + half * 8;
      out[r]      = 1.0f / (1.0f + __expf(-pA[v]));
      out[r + 16] = 1.0f / (1.0f + __expf(-pB[v]));
    }
  }
}

extern "C" void kernel_launch(void* const* d_in, const int* in_sizes, int n_in,
                              void* d_out, int out_size, void* d_ws, size_t ws_size,
                              hipStream_t stream) {
  (void)in_sizes; (void)n_in; (void)out_size;
  const float* features = (const float*)d_in[0];
  const float* patterns = (const float*)d_in[1];
  const float* Wout     = (const float*)d_in[2];
  const float* bout     = (const float*)d_in[3];
  float* out = (float*)d_out;

  const size_t WS_NEEDED = 2u * (size_t)QPAD * DDIM * sizeof(unsigned short);
  if (d_ws != nullptr && ws_size >= WS_NEEDED) {
    unsigned short* wsRow = (unsigned short*)d_ws;
    unsigned short* wsCol = wsRow + (size_t)QPAD * DDIM;
    hipLaunchKernelGGL(hopfield_prep_kernel, dim3(QPAD * 64 / 256), dim3(256), 0,
                       stream, patterns, wsRow, wsCol);
    hipLaunchKernelGGL(hopfield_ws_kernel, dim3(BATCH / ROWS_PER_WG), dim3(BLOCK),
                       0, stream, features, wsRow, wsCol, Wout, bout, out);
  } else {
    hipLaunchKernelGGL(hopfield_fallback_kernel, dim3(BATCH / ROWS_PER_WG),
                       dim3(BLOCK), 0, stream, features, patterns, Wout, bout, out);
  }
}
